// MultiheadAttention_33715493274048
// MI455X (gfx1250) — compile-verified, matrix operand order checked by reading, not run on hardware
//
#include <hip/hip_runtime.h>
#include <hip/hip_bf16.h>

// ---------------- problem constants ----------------
#define NB 2
#define NS 2048
#define ND 1024
#define NH 16
#define HD 64

typedef _Float16 f16;
typedef __attribute__((ext_vector_type(16))) _Float16 v16h;
typedef __attribute__((ext_vector_type(8)))  _Float16 v8h;
typedef __attribute__((ext_vector_type(8)))  float    v8f;
typedef __attribute__((ext_vector_type(4)))  int      v4i;

union Frag16 { v16h v; v8h h[2]; };

// ---------------- CDNA5 async-to-LDS helpers ----------------
// clang-22 signature: (v4i addrspace(1)* src, v4i addrspace(3)* dst, imm, imm)
typedef __attribute__((address_space(1))) v4i gv4i;
typedef __attribute__((address_space(3))) v4i lv4i;

__device__ __forceinline__ void async_copy_b128(const f16* g, f16* l) {
#if __has_builtin(__builtin_amdgcn_global_load_async_to_lds_b128)
    __builtin_amdgcn_global_load_async_to_lds_b128(
        (gv4i*)(unsigned long long)g,
        (lv4i*)(unsigned int)(unsigned long long)l, 0, 0);
#else
    unsigned int       loff  = (unsigned int)(unsigned long long)l;
    unsigned long long gaddr = (unsigned long long)g;
    asm volatile("global_load_async_to_lds_b128 %0, %1, off"
                 :: "v"(loff), "v"(gaddr) : "memory");
#endif
}

__device__ __forceinline__ void wait_async0() {
#if __has_builtin(__builtin_amdgcn_s_wait_asynccnt)
    __builtin_amdgcn_s_wait_asynccnt(0);
#else
    asm volatile("s_wait_asynccnt 0" ::: "memory");
#endif
}

// ---------------- workspace layout (bytes) ----------------
static constexpr size_t XE = (size_t)NB * NS * ND;   // 4,194,304 elems
static constexpr size_t WE = (size_t)ND * ND;        // 1,048,576 elems
static constexpr size_t OFF_XH = 0;                  // x as f16
static constexpr size_t OFF_WQ = OFF_XH + XE * 2;
static constexpr size_t OFF_WK = OFF_WQ + WE * 2;
static constexpr size_t OFF_WV = OFF_WK + WE * 2;
static constexpr size_t OFF_Q  = OFF_WV + WE * 2;    // q  [B,H,S,hd] f16
static constexpr size_t OFF_K  = OFF_Q  + XE * 2;    // k  [B,H,S,hd] f16
static constexpr size_t OFF_VT = OFF_K  + XE * 2;    // vT [B,H,hd,S] f16
static constexpr size_t OFF_AO = OFF_VT + XE * 2;    // attn out [B,S,D] f32

// ---------------- 1) f32 -> f16 convert ----------------
__global__ __launch_bounds__(256)
void cvt_f32_f16_kernel(const float* __restrict__ in, f16* __restrict__ out, int n) {
    int i = blockIdx.x * 256 + threadIdx.x;
    if (i < n) out[i] = (f16)in[i];
}

// ---------------- 2) fused QKV GEMM ----------------
// C[M=4096, N=3072] = Xh[4096,1024] @ Wsel^T, Wsel row-major [1024,1024].
// Block: 256 threads (8 waves). Tile: M=128 (8 waves x 16), N=64, K-step=32.
// Weight tiles double-buffered in LDS via async-to-LDS DMA.
__global__ __launch_bounds__(256)
void qkv_gemm_kernel(const f16* __restrict__ xh,
                     const f16* __restrict__ wqh, const f16* __restrict__ wkh,
                     const f16* __restrict__ wvh,
                     const float* __restrict__ bq, const float* __restrict__ bk,
                     const float* __restrict__ bv,
                     f16* __restrict__ qout, f16* __restrict__ kout,
                     f16* __restrict__ vTout)
{
    constexpr int KP = 40;                  // LDS row pitch in halves (80B, 16B aligned)
    __shared__ f16 ldsB[2][64 * KP];

    const int tid  = threadIdx.x;
    const int wave = tid >> 5;
    const int lane = tid & 31;
    const int hi   = lane >> 4;             // half-wave select (A/B frag K interleave)
    const int lr   = lane & 15;

    const int nbase = blockIdx.y * 64;      // 0..3071 across {wq,wk,wv}
    const int mat   = nbase >> 10;          // 0=q 1=k 2=v (uniform per block)
    const int n0    = nbase & 1023;         // column base within matrix
    const f16*   w    = (mat == 0) ? wqh : (mat == 1) ? wkh : wvh;
    const float* bias = (mat == 0) ? bq  : (mat == 1) ? bk  : bv;

    const int m0 = blockIdx.x * 128 + wave * 16;

    // per-thread slice of the 64x32 weight tile: one b128 each
    const int wr = tid >> 2;                // 0..63 (row n within tile)
    const int wc = (tid & 3) * 8;           // 0,8,16,24 halves

    v8f acc[4] = {};                        // 16x64 f32 accumulators per wave

    // prologue: stage tile 0
    async_copy_b128(w + (size_t)(n0 + wr) * ND + wc, &ldsB[0][wr * KP + wc]);

    constexpr int NSTEP = ND / 32;          // 32 K-steps
    for (int step = 0; step < NSTEP; ++step) {
        const int k0  = step * 32;
        const int buf = step & 1;

        wait_async0();                      // my slice of tile `step` arrived
        __syncthreads();                    // everyone's slice arrived

        if (step + 1 < NSTEP)               // prefetch next tile into other buffer
            async_copy_b128(w + (size_t)(n0 + wr) * ND + (k0 + 32) + wc,
                            &ldsB[buf ^ 1][wr * KP + wc]);

        // A fragment (16x32) straight from global: ISA 16-bit A layout
        Frag16 a;
        {
            const f16* arow = xh + (size_t)(m0 + lr) * ND + k0 + hi * 8;
            a.h[0] = *(const v8h*)(arow);
            a.h[1] = *(const v8h*)(arow + 16);
        }

        // 4 B fragments from LDS (N groups of 16), accumulate
        #pragma unroll
        for (int g = 0; g < 4; ++g) {
            Frag16 b;
            const f16* brow = &ldsB[buf][(g * 16 + lr) * KP + hi * 8];
            b.h[0] = *(const v8h*)(brow);
            b.h[1] = *(const v8h*)(brow + 16);
            acc[g] = __builtin_amdgcn_wmma_f32_16x16x32_f16(
                false, a.v, false, b.v, (short)0, acc[g], false, false);
        }
        // no trailing barrier: next overwrite of ldsB[buf] happens only after
        // the next iteration's wait+barrier, by which time all reads completed
    }

    // epilogue: +bias, convert to f16, scatter into head-major layouts
    #pragma unroll
    for (int g = 0; g < 4; ++g) {
        int e  = n0 + g * 16 + lr;           // 0..1023 within matrix
        float bb = bias[e];
        int h = e >> 6, d = e & 63;
        #pragma unroll
        for (int j = 0; j < 8; ++j) {
            int m = m0 + j + hi * 8;         // global row (b*S + s)
            int b = m >> 11, s = m & (NS - 1);
            f16 hv = (f16)(acc[g][j] + bb);
            if (mat == 0)
                qout[(((size_t)(b * NH + h) * NS) + s) * HD + d] = hv;
            else if (mat == 1)
                kout[(((size_t)(b * NH + h) * NS) + s) * HD + d] = hv;
            else
                vTout[(((size_t)(b * NH + h) * HD) + d) * NS + s] = hv;
        }
    }
}

// ---------------- 3) flash attention ----------------
// Grid: (S/128, B*H). Block 256 = 8 waves; each wave owns 16 query rows.
// K/V tiles staged once per block (shared by all 8 waves) in double-buffered
// LDS via async-to-LDS DMA; P re-layout staged through per-wave LDS.
__global__ __launch_bounds__(256)
void attn_kernel(const f16* __restrict__ qbuf, const f16* __restrict__ kbuf,
                 const f16* __restrict__ vT, float* __restrict__ obuf)
{
    constexpr int KTP = 72;                  // K tile pitch (64+8 halves)
    constexpr int VTP = 40;                  // V tile pitch (32+8 halves)
    constexpr int PP  = 40;                  // per-wave P tile pitch
    __shared__ f16 ldsK[2][32 * KTP];        // 32 keys x 64 dims
    __shared__ f16 ldsV[2][64 * VTP];        // 64 dims x 32 keys (vT rows)
    __shared__ f16 ldsP[8][16 * PP];

    const int tid  = threadIdx.x;
    const int wave = tid >> 5;
    const int lane = tid & 31;
    const int hi   = lane >> 4;
    const int lr   = lane & 15;

    const int bh = blockIdx.y;               // b*NH + h
    const int q0 = blockIdx.x * 128 + wave * 16;

    const f16* qp = qbuf + (size_t)bh * NS * HD;
    const f16* kp = kbuf + (size_t)bh * NS * HD;
    const f16* vp = vT   + (size_t)bh * HD * NS;
    f16* myP = &ldsP[wave][0];

    // per-thread async slices: K tile row tid>>3, chunk (tid&7)*8;
    //                          V tile row tid>>2, chunk (tid&3)*8
    const int kr = tid >> 3, kc = (tid & 7) * 8;
    const int vr = tid >> 2, vc = (tid & 3) * 8;

    // Q A-fragments (16 rows x hd=64 -> 2 frags), kept in registers
    Frag16 qa[2];
    #pragma unroll
    for (int t = 0; t < 2; ++t) {
        const f16* row = qp + (size_t)(q0 + lr) * HD + t * 32 + hi * 8;
        qa[t].h[0] = *(const v8h*)row;
        qa[t].h[1] = *(const v8h*)(row + 16);
    }

    v8f acc[4] = {};                          // 16 x 64 output accumulator
    float mrow[8], lsum[8];
    #pragma unroll
    for (int j = 0; j < 8; ++j) { mrow[j] = -1e30f; lsum[j] = 0.0f; }

    // prologue: stage key-block 0
    async_copy_b128(kp + (size_t)kr * HD + kc, &ldsK[0][kr * KTP + kc]);
    async_copy_b128(vp + (size_t)vr * NS + vc, &ldsV[0][vr * VTP + vc]);

    constexpr int NKB = NS / 32;              // 64 key blocks
    for (int kb = 0; kb < NKB; ++kb) {
        const int k0  = kb * 32;
        const int buf = kb & 1;

        wait_async0();
        __syncthreads();

        if (kb + 1 < NKB) {
            async_copy_b128(kp + (size_t)(k0 + 32 + kr) * HD + kc,
                            &ldsK[buf ^ 1][kr * KTP + kc]);
            async_copy_b128(vp + (size_t)vr * NS + (k0 + 32) + vc,
                            &ldsV[buf ^ 1][vr * VTP + vc]);
        }

        // scores 16x32 = Q(16x64) @ K_blk^T : two 16x16 frags, K split in 2
        v8f sc[2] = {};
        #pragma unroll
        for (int kh = 0; kh < 2; ++kh) {      // key sub-group of 16
            #pragma unroll
            for (int t = 0; t < 2; ++t) {     // head-dim chunk
                Frag16 bfr;
                const f16* row = &ldsK[buf][(kh * 16 + lr) * KTP + t * 32 + hi * 8];
                bfr.h[0] = *(const v8h*)row;
                bfr.h[1] = *(const v8h*)(row + 16);
                sc[kh] = __builtin_amdgcn_wmma_f32_16x16x32_f16(
                    false, qa[t].v, false, bfr.v, (short)0, sc[kh], false, false);
            }
        }

        // online softmax (f32); row r = j + 8*hi lives across one 16-lane group
        float fac[8];
        #pragma unroll
        for (int j = 0; j < 8; ++j) {
            float s0 = sc[0][j] * 0.125f;     // 1/sqrt(64)
            float s1 = sc[1][j] * 0.125f;
            float pm = fmaxf(s0, s1);
            pm = fmaxf(pm, __shfl_xor(pm, 1, 32));
            pm = fmaxf(pm, __shfl_xor(pm, 2, 32));
            pm = fmaxf(pm, __shfl_xor(pm, 4, 32));
            pm = fmaxf(pm, __shfl_xor(pm, 8, 32));
            float mn = fmaxf(mrow[j], pm);
            float f  = __expf(mrow[j] - mn);
            float p0 = __expf(s0 - mn);
            float p1 = __expf(s1 - mn);
            float ps = p0 + p1;
            ps += __shfl_xor(ps, 1, 32);
            ps += __shfl_xor(ps, 2, 32);
            ps += __shfl_xor(ps, 4, 32);
            ps += __shfl_xor(ps, 8, 32);
            mrow[j] = mn;
            lsum[j] = lsum[j] * f + ps;
            fac[j]  = f;
            // stage P (f16) into per-wave LDS tile, C-layout -> row-major
            int r = j + hi * 8;
            myP[r * PP + lr]      = (f16)p0;
            myP[r * PP + 16 + lr] = (f16)p1;
        }
        #pragma unroll
        for (int c = 0; c < 4; ++c)
            #pragma unroll
            for (int j = 0; j < 8; ++j) acc[c][j] *= fac[j];

        // same-wave DS ops are in-order; wait for stores, then A-frag reload
        asm volatile("s_wait_dscnt 0" ::: "memory");
        Frag16 pa;
        {
            const f16* row = &myP[lr * PP + hi * 8];
            pa.h[0] = *(const v8h*)row;
            pa.h[1] = *(const v8h*)(row + 16);
        }

        // out += P(16x32) @ V_blk(32x64); staged vT rows are contiguous B-frags
        #pragma unroll
        for (int c = 0; c < 4; ++c) {
            Frag16 bfr;
            const f16* row = &ldsV[buf][(c * 16 + lr) * VTP + hi * 8];
            bfr.h[0] = *(const v8h*)row;
            bfr.h[1] = *(const v8h*)(row + 16);
            acc[c] = __builtin_amdgcn_wmma_f32_16x16x32_f16(
                false, pa.v, false, bfr.v, (short)0, acc[c], false, false);
        }
    }

    // normalize and write f32 [B,S,D]
    int b = bh >> 4, h = bh & 15;
    #pragma unroll
    for (int j = 0; j < 8; ++j) {
        float inv = 1.0f / lsum[j];
        int s = q0 + j + hi * 8;
        float* orow = obuf + ((size_t)b * NS + s) * ND + h * HD;
        #pragma unroll
        for (int c = 0; c < 4; ++c) orow[c * 16 + lr] = acc[c][j] * inv;
    }
}

// ---------------- 4) residual + LayerNorm ----------------
__global__ __launch_bounds__(256)
void resid_ln_kernel(const float* __restrict__ attn, const float* __restrict__ x,
                     const float* __restrict__ gamma, const float* __restrict__ beta,
                     float* __restrict__ out)
{
    __shared__ float s1[8], s2[8];
    const int row = blockIdx.x;
    const int tid = threadIdx.x;
    const int wave = tid >> 5, lane = tid & 31;
    const float* a  = attn + (size_t)row * ND;
    const float* xr = x    + (size_t)row * ND;

    float v[4], sum = 0.0f, sq = 0.0f;
    #pragma unroll
    for (int i = 0; i < 4; ++i) {
        float t = a[tid + i * 256] + xr[tid + i * 256];
        v[i] = t; sum += t; sq += t * t;
    }
    #pragma unroll
    for (int off = 1; off < 32; off <<= 1) {
        sum += __shfl_xor(sum, off, 32);
        sq  += __shfl_xor(sq,  off, 32);
    }
    if (lane == 0) { s1[wave] = sum; s2[wave] = sq; }
    __syncthreads();
    if (wave == 0) {
        float a1 = (lane < 8) ? s1[lane] : 0.0f;
        float a2 = (lane < 8) ? s2[lane] : 0.0f;
        #pragma unroll
        for (int off = 1; off < 8; off <<= 1) {
            a1 += __shfl_xor(a1, off, 32);
            a2 += __shfl_xor(a2, off, 32);
        }
        if (lane == 0) { s1[0] = a1; s2[0] = a2; }
    }
    __syncthreads();
    float mean = s1[0] * (1.0f / ND);
    float var  = s2[0] * (1.0f / ND) - mean * mean;
    float rstd = rsqrtf(var + 1e-5f);
    float* orow = out + (size_t)row * ND;
    #pragma unroll
    for (int i = 0; i < 4; ++i) {
        int c = tid + i * 256;
        orow[c] = (v[i] - mean) * rstd * gamma[c] + beta[c];
    }
}

// ---------------- launch ----------------
extern "C" void kernel_launch(void* const* d_in, const int* in_sizes, int n_in,
                              void* d_out, int out_size, void* d_ws, size_t ws_size,
                              hipStream_t stream) {
    (void)in_sizes; (void)n_in; (void)out_size; (void)ws_size;
    const float* x     = (const float*)d_in[0];
    const float* wq    = (const float*)d_in[1];
    const float* bq    = (const float*)d_in[2];
    const float* wk    = (const float*)d_in[3];
    const float* bk    = (const float*)d_in[4];
    const float* wv    = (const float*)d_in[5];
    const float* bv    = (const float*)d_in[6];
    const float* gamma = (const float*)d_in[7];
    const float* beta  = (const float*)d_in[8];

    char* ws = (char*)d_ws;
    f16*   xh   = (f16*)(ws + OFF_XH);
    f16*   wqh  = (f16*)(ws + OFF_WQ);
    f16*   wkh  = (f16*)(ws + OFF_WK);
    f16*   wvh  = (f16*)(ws + OFF_WV);
    f16*   qb   = (f16*)(ws + OFF_Q);
    f16*   kb   = (f16*)(ws + OFF_K);
    f16*   vtb  = (f16*)(ws + OFF_VT);
    float* aout = (float*)(ws + OFF_AO);

    cvt_f32_f16_kernel<<<(int)((XE + 255) / 256), 256, 0, stream>>>(x,  xh,  (int)XE);
    cvt_f32_f16_kernel<<<(int)((WE + 255) / 256), 256, 0, stream>>>(wq, wqh, (int)WE);
    cvt_f32_f16_kernel<<<(int)((WE + 255) / 256), 256, 0, stream>>>(wk, wkh, (int)WE);
    cvt_f32_f16_kernel<<<(int)((WE + 255) / 256), 256, 0, stream>>>(wv, wvh, (int)WE);

    qkv_gemm_kernel<<<dim3(32, 48), 256, 0, stream>>>(
        xh, wqh, wkh, wvh, bq, bk, bv, qb, kb, vtb);

    attn_kernel<<<dim3(NS / 128, NB * NH), 256, 0, stream>>>(qb, kb, vtb, aout);

    resid_ln_kernel<<<NB * NS, 256, 0, stream>>>(aout, x, gamma, beta, (float*)d_out);
}
